// MultiHeadAttention_81896436400192
// MI455X (gfx1250) — matrix operand order repaired and checked by reading, not compiled, not run on hardware
//
#include <hip/hip_runtime.h>

// ---------------- problem constants ----------------
#define B_SZ   2
#define S_LEN  2048
#define D_DIM  1024
#define H_NUM  16
#define DH     64
#define BS     (B_SZ * S_LEN)     // 4096 token rows

// ---------------- WMMA types ----------------
typedef __bf16 v16bf __attribute__((ext_vector_type(16)));
typedef __bf16 v2bf  __attribute__((ext_vector_type(2)));
typedef float  v8f   __attribute__((ext_vector_type(8)));

union FragB {               // 16 bf16 = 32 bytes = two 16B chunks
    v16bf v;
    uint4 q[2];
};

__device__ __forceinline__ v8f wmma_bf16(const FragB& a, const FragB& b, v8f c) {
    return __builtin_amdgcn_wmma_f32_16x16x32_bf16(
        /*neg_a=*/false, a.v, /*neg_b=*/false, b.v,
        /*c_mod=*/(short)0, c, /*reuse_a=*/false, /*reuse_b=*/false);
}

// hardware f32->bf16 (RNE) -- lowers to v_cvt_*bf16* instead of add3/mov chains
__device__ __forceinline__ unsigned short f32_bf16(float f) {
    return __builtin_bit_cast(unsigned short, (__bf16)f);
}
__device__ __forceinline__ unsigned pack2_bf16(float a, float b) {
    v2bf p; p[0] = (__bf16)a; p[1] = (__bf16)b;
    return __builtin_bit_cast(unsigned, p);
}
__device__ __forceinline__ uint4 pack8_bf16(float4 f0, float4 f1) {
    uint4 q;
    q.x = pack2_bf16(f0.x, f0.y); q.y = pack2_bf16(f0.z, f0.w);
    q.z = pack2_bf16(f1.x, f1.y); q.w = pack2_bf16(f1.z, f1.w);
    return q;
}

// ---------------- gfx1250 async copy: global -> LDS (ASYNCcnt) ----------------
__device__ __forceinline__ unsigned lds_off32(const void* p) {
    // flat pointers to LDS carry the aperture in the high 32 bits only;
    // low 32 bits are the LDS byte offset (ISA 10.2 aperture decode)
    return (unsigned)(unsigned long long)p;
}
__device__ __forceinline__ void async_load_b128(unsigned lds_addr, const void* gptr) {
    unsigned long long ga = (unsigned long long)gptr;
    asm volatile("global_load_async_to_lds_b128 %0, %1, off"
                 :: "v"(lds_addr), "v"(ga) : "memory");
}
__device__ __forceinline__ void wait_async0() {
    asm volatile("s_wait_asynccnt 0x0" ::: "memory");
}

// =====================================================================
// Kernel 1: fused Q/K/V projection.  C = X @ W^T + b, emitted as bf16.
// Q,K -> [B,H,S,DH]; V -> [B,H,DH,S] (transposed so attention's PV
// B-fragments are k-contiguous).  grid: (BS/128, D/128, 3), 256 thr.
// Score scale 1/sqrt(DH) = 1/8 folded into Q.
// =====================================================================
__global__ __launch_bounds__(256)
void qkv_proj_kernel(const float* __restrict__ Xq, const float* __restrict__ Xk,
                     const float* __restrict__ Xv,
                     const float* __restrict__ Wq, const float* __restrict__ Wk,
                     const float* __restrict__ Wv,
                     const float* __restrict__ bq, const float* __restrict__ bk,
                     const float* __restrict__ bv,
                     unsigned short* __restrict__ Qw, unsigned short* __restrict__ Kw,
                     unsigned short* __restrict__ Vw)
{
    const int which = blockIdx.z;
    const float* X    = (which == 0) ? Xq : (which == 1) ? Xk : Xv;
    const float* W    = (which == 0) ? Wq : (which == 1) ? Wk : Wv;
    const float* bias = (which == 0) ? bq : (which == 1) ? bk : bv;
    unsigned short* Out = (which == 0) ? Qw : (which == 1) ? Kw : Vw;
    const float scale = (which == 0) ? 0.125f : 1.0f;

    __shared__ __align__(16) unsigned short At[128][32];  // [m][k] bf16
    __shared__ __align__(16) unsigned short Bt[128][32];  // [n][k] bf16

    const int tid  = threadIdx.x;
    const int lane = tid & 31;
    const int wave = tid >> 5;
    const int m0 = blockIdx.x * 128;
    const int n0 = blockIdx.y * 128;

    v8f acc[8];
    #pragma unroll
    for (int i = 0; i < 8; ++i) { v8f z = {0,0,0,0,0,0,0,0}; acc[i] = z; }

    const int ldRow = tid >> 1;          // 0..127
    const int ldSeg = (tid & 1) * 16;    // 0 or 16 (fp32 elems)
    const float* xRow = X + (size_t)(m0 + ldRow) * D_DIM + ldSeg;
    const float* wRow = W + (size_t)(n0 + ldRow) * D_DIM + ldSeg;

    for (int kb = 0; kb < D_DIM / 32; ++kb) {
        const int k0 = kb * 32;
        if (kb + 1 < D_DIM / 32) {                 // global_prefetch_b8 next tiles
            __builtin_prefetch(xRow + k0 + 32, 0, 1);
            __builtin_prefetch(wRow + k0 + 32, 0, 1);
        }
        // ---- stage X tile (hw cvt fp32 -> bf16) ----
        {
            const float4* s4 = (const float4*)(xRow + k0);
            float4 f0 = s4[0], f1 = s4[1], f2 = s4[2], f3 = s4[3];
            *(uint4*)&At[ldRow][ldSeg]       = pack8_bf16(f0, f1);
            *(uint4*)(&At[ldRow][ldSeg] + 8) = pack8_bf16(f2, f3);
        }
        // ---- stage W tile ----
        {
            const float4* s4 = (const float4*)(wRow + k0);
            float4 f0 = s4[0], f1 = s4[1], f2 = s4[2], f3 = s4[3];
            *(uint4*)&Bt[ldRow][ldSeg]       = pack8_bf16(f0, f1);
            *(uint4*)(&Bt[ldRow][ldSeg] + 8) = pack8_bf16(f2, f3);
        }
        __syncthreads();

        // ---- WMMA: each wave owns 16 rows x 128 cols ----
        FragB a;
        {
            const int m    = wave * 16 + (lane & 15);
            const int koff = (lane >> 4) * 16;            // A layout: 16 contiguous k
            a.q[0] = *(const uint4*)&At[m][koff];
            a.q[1] = *(const uint4*)&At[m][koff + 8];
        }
        #pragma unroll
        for (int nt = 0; nt < 8; ++nt) {
            FragB b;
            const int n  = nt * 16 + (lane & 15);
            const int kh = (lane >> 4) * 16;              // B layout: 16 contiguous k
            b.q[0] = *(const uint4*)&Bt[n][kh];
            b.q[1] = *(const uint4*)(&Bt[n][kh] + 8);
            acc[nt] = wmma_bf16(a, b, acc[nt]);
        }
        __syncthreads();
    }

    // ---- epilogue: +bias, scale, scatter ----
    #pragma unroll
    for (int nt = 0; nt < 8; ++nt) {
        const int n = n0 + nt * 16 + (lane & 15);
        const float bb = bias[n];
        const int h = n >> 6, d = n & 63;
        #pragma unroll
        for (int r = 0; r < 8; ++r) {
            const int m = m0 + wave * 16 + r + ((lane >> 4) << 3);
            const int bi = m >> 11;           // / S_LEN
            const int s  = m & (S_LEN - 1);
            const unsigned short bits = f32_bf16((acc[nt][r] + bb) * scale);
            if (which == 2)   // V transposed: [B,H,DH,S]
                Out[(((size_t)bi * H_NUM + h) * DH + d) * S_LEN + s] = bits;
            else              // Q,K: [B,H,S,DH]
                Out[(((size_t)bi * H_NUM + h) * S_LEN + s) * DH + d] = bits;
        }
    }
}

// =====================================================================
// Kernel 2: flash attention.  grid: (B*H, S/128), 256 threads (8 waves).
// K/V tiles staged by global_load_async_to_lds_b128; online softmax;
// P round-trips per-wave LDS for C-layout -> A-layout.
// =====================================================================
__global__ __launch_bounds__(256)
void attn_kernel(const unsigned short* __restrict__ Qw,
                 const unsigned short* __restrict__ Kw,
                 const unsigned short* __restrict__ Vw,
                 unsigned short* __restrict__ Att)
{
    __shared__ __align__(16) unsigned short Kt[64][64];       // [key][dh]
    __shared__ __align__(16) unsigned short Vt[64][64];       // [dh][key]
    __shared__ __align__(16) unsigned short Pb[8][16][64];    // per-wave P tile

    const int tid  = threadIdx.x;
    const int lane = tid & 31;
    const int wave = tid >> 5;
    const int col  = lane & 15;
    const int half = lane >> 4;

    const int bh = blockIdx.x;                // b*H + h
    const int q0 = blockIdx.y * 128;
    const size_t head = (size_t)bh * S_LEN * DH;   // same size for Q/K and V^T

    // Q A-fragments live in registers for the whole kernel
    FragB aq[2];
    {
        const int m = q0 + wave * 16 + col;
        const unsigned short* qrow = Qw + head + (size_t)m * DH;
        const int koff = half * 16;
        #pragma unroll
        for (int ks = 0; ks < 2; ++ks) {
            aq[ks].q[0] = *(const uint4*)(qrow + ks * 32 + koff);
            aq[ks].q[1] = *(const uint4*)(qrow + ks * 32 + koff + 8);
        }
    }

    float mrow[8], lrow[8];
    v8f oacc[4];
    #pragma unroll
    for (int r = 0; r < 8; ++r) { mrow[r] = -1e30f; lrow[r] = 0.0f; }
    #pragma unroll
    for (int i = 0; i < 4; ++i) { v8f z = {0,0,0,0,0,0,0,0}; oacc[i] = z; }

    const unsigned ldsK = lds_off32(&Kt[0][0]);
    const int vr = tid >> 2;            // 0..63 (dh row of V^T tile)
    const int vj = (tid & 3) * 2;       // chunk pair within 128B row
    const unsigned ldsV = lds_off32(&Vt[vr][0]) + (unsigned)vj * 16u;

    for (int kb = 0; kb < S_LEN / 64; ++kb) {
        // ---- async stage: K tile is 8KB contiguous, V^T rows are 128B runs ----
        {
            const unsigned short* kbase = Kw + head + (size_t)kb * 64 * DH;
            async_load_b128(ldsK + (unsigned)tid * 16u,        kbase + (size_t)tid * 8);
            async_load_b128(ldsK + (unsigned)(tid + 256) * 16u, kbase + (size_t)(tid + 256) * 8);
            const unsigned short* vrow = Vw + head + (size_t)vr * S_LEN + kb * 64 + vj * 8;
            async_load_b128(ldsV,       vrow);
            async_load_b128(ldsV + 16u, vrow + 8);
        }
        wait_async0();
        __syncthreads();

        // ---- S = Q K^T  (16x64 per wave) ----
        v8f sacc[4];
        #pragma unroll
        for (int i = 0; i < 4; ++i) { v8f z = {0,0,0,0,0,0,0,0}; sacc[i] = z; }
        #pragma unroll
        for (int nt = 0; nt < 4; ++nt) {
            #pragma unroll
            for (int ks = 0; ks < 2; ++ks) {
                FragB b;
                const int key = nt * 16 + col;
                const int kh  = ks * 32 + half * 16;
                b.q[0] = *(const uint4*)&Kt[key][kh];
                b.q[1] = *(const uint4*)(&Kt[key][kh] + 8);
                sacc[nt] = wmma_bf16(aq[ks], b, sacc[nt]);
            }
        }

        // ---- online softmax (rows live in 16-lane halves) ----
        #pragma unroll
        for (int r = 0; r < 8; ++r) {
            float mx = fmaxf(fmaxf(sacc[0][r], sacc[1][r]),
                             fmaxf(sacc[2][r], sacc[3][r]));
            #pragma unroll
            for (int off = 1; off < 16; off <<= 1)
                mx = fmaxf(mx, __shfl_xor(mx, off, 32));
            const float mnew  = fmaxf(mrow[r], mx);
            const float alpha = __expf(mrow[r] - mnew);
            mrow[r] = mnew;
            float rs = 0.0f;
            #pragma unroll
            for (int nt = 0; nt < 4; ++nt) {
                const float p = __expf(sacc[nt][r] - mnew);
                sacc[nt][r] = p;
                rs += p;
            }
            #pragma unroll
            for (int off = 1; off < 16; off <<= 1)
                rs += __shfl_xor(rs, off, 32);
            lrow[r] = lrow[r] * alpha + rs;
            #pragma unroll
            for (int dt = 0; dt < 4; ++dt) oacc[dt][r] *= alpha;
        }

        // ---- P: accumulator layout -> per-wave LDS (bf16) ----
        #pragma unroll
        for (int nt = 0; nt < 4; ++nt)
            #pragma unroll
            for (int r = 0; r < 8; ++r)
                Pb[wave][r + half * 8][nt * 16 + col] = f32_bf16(sacc[nt][r]);

        // ---- O += P V ----
        #pragma unroll
        for (int ks = 0; ks < 2; ++ks) {
            FragB ap;
            const int koff = ks * 32 + half * 16;
            ap.q[0] = *(const uint4*)&Pb[wave][col][koff];
            ap.q[1] = *(const uint4*)(&Pb[wave][col][koff] + 8);
            #pragma unroll
            for (int dt = 0; dt < 4; ++dt) {
                FragB b;
                const int dh = dt * 16 + col;
                const int kh = ks * 32 + half * 16;
                b.q[0] = *(const uint4*)&Vt[dh][kh];
                b.q[1] = *(const uint4*)(&Vt[dh][kh] + 8);
                oacc[dt] = wmma_bf16(ap, b, oacc[dt]);
            }
        }
        __syncthreads();
    }

    // ---- epilogue: O * (1/l), scatter back to [B,S,D] bf16 ----
    const int bb = bh >> 4, hh = bh & 15;
    #pragma unroll
    for (int r = 0; r < 8; ++r) {
        const float rl = 1.0f / lrow[r];
        const int s = q0 + wave * 16 + r + half * 8;
        #pragma unroll
        for (int dt = 0; dt < 4; ++dt) {
            const int c = hh * 64 + dt * 16 + col;
            Att[((size_t)(bb * S_LEN + s)) * D_DIM + c] = f32_bf16(oacc[dt][r] * rl);
        }
    }
}

// =====================================================================
// Kernel 3: output projection.  out = Att(bf16) @ Wo^T + bo, fp32 out.
// bf16 activations staged via async LDS copy.  grid: (BS/128, D/128).
// =====================================================================
__global__ __launch_bounds__(256)
void out_proj_kernel(const unsigned short* __restrict__ Xh,
                     const float* __restrict__ W,
                     const float* __restrict__ bias,
                     float* __restrict__ Out)
{
    __shared__ __align__(16) unsigned short At[128][32];
    __shared__ __align__(16) unsigned short Bt[128][32];

    const int tid  = threadIdx.x;
    const int lane = tid & 31;
    const int wave = tid >> 5;
    const int m0 = blockIdx.x * 128;
    const int n0 = blockIdx.y * 128;

    v8f acc[8];
    #pragma unroll
    for (int i = 0; i < 8; ++i) { v8f z = {0,0,0,0,0,0,0,0}; acc[i] = z; }

    const int ldRow = tid >> 1;
    const int ldSeg = (tid & 1) * 16;
    const float* wRow = W + (size_t)(n0 + ldRow) * D_DIM + ldSeg;

    // async staging split: 512 16B chunks of the A tile over 256 threads
    const int ar = tid >> 1;                 // 0..127 rows, 2 chunks each
    const int aj = (tid & 1) * 2;            // chunk pair (of 4 per 64B row)
    const unsigned ldsA = lds_off32(&At[ar][0]) + (unsigned)aj * 16u;

    for (int kb = 0; kb < D_DIM / 32; ++kb) {
        const int k0 = kb * 32;
        if (kb + 1 < D_DIM / 32)
            __builtin_prefetch(wRow + k0 + 32, 0, 1);
        // ---- async stage bf16 activations ----
        {
            const unsigned short* arow = Xh + (size_t)(m0 + ar) * D_DIM + k0 + aj * 8;
            async_load_b128(ldsA,       arow);
            async_load_b128(ldsA + 16u, arow + 8);
        }
        // ---- stage Wo tile (hw cvt fp32 -> bf16) ----
        {
            const float4* s4 = (const float4*)(wRow + k0);
            float4 f0 = s4[0], f1 = s4[1], f2 = s4[2], f3 = s4[3];
            *(uint4*)&Bt[ldRow][ldSeg]       = pack8_bf16(f0, f1);
            *(uint4*)(&Bt[ldRow][ldSeg] + 8) = pack8_bf16(f2, f3);
        }
        wait_async0();
        __syncthreads();

        FragB a;
        {
            const int m    = wave * 16 + (lane & 15);
            const int koff = (lane >> 4) * 16;
            a.q[0] = *(const uint4*)&At[m][koff];
            a.q[1] = *(const uint4*)&At[m][koff + 8];
        }
        #pragma unroll
        for (int nt = 0; nt < 8; ++nt) {
            FragB b;
            const int n  = nt * 16 + (lane & 15);
            const int kh = (lane >> 4) * 16;
            b.q[0] = *(const uint4*)&Bt[n][kh];
            b.q[1] = *(const uint4*)(&Bt[n][kh] + 8);
            acc[nt] = wmma_bf16(a, b, acc[nt]);
        }
        __syncthreads();
    }

    #pragma unroll
    for (int nt = 0; nt < 8; ++nt) {
        const int n = n0 + nt * 16 + (lane & 15);
        const float bb = bias[n];
        #pragma unroll
        for (int r = 0; r < 8; ++r) {
            const int m = m0 + wave * 16 + r + ((lane >> 4) << 3);
            Out[(size_t)m * D_DIM + n] = acc[nt][r] + bb;
        }
    }
}

// =====================================================================
// launcher
// =====================================================================
extern "C" void kernel_launch(void* const* d_in, const int* in_sizes, int n_in,
                              void* d_out, int out_size, void* d_ws, size_t ws_size,
                              hipStream_t stream) {
    (void)in_sizes; (void)n_in; (void)out_size; (void)ws_size;

    const float* query  = (const float*)d_in[0];
    const float* keys   = (const float*)d_in[1];
    const float* values = (const float*)d_in[2];
    const float* Wq = (const float*)d_in[3];
    const float* bq = (const float*)d_in[4];
    const float* Wk = (const float*)d_in[5];
    const float* bk = (const float*)d_in[6];
    const float* Wv = (const float*)d_in[7];
    const float* bv = (const float*)d_in[8];
    const float* Wo = (const float*)d_in[9];
    const float* bo = (const float*)d_in[10];
    float* out = (float*)d_out;

    const size_t elems = (size_t)BS * D_DIM;    // 4M bf16 elements each
    unsigned short* Qw  = (unsigned short*)d_ws;
    unsigned short* Kw  = Qw + elems;
    unsigned short* Vw  = Kw + elems;           // stored as [B,H,DH,S]
    unsigned short* Att = Vw + elems;

    dim3 blk(256);
    qkv_proj_kernel<<<dim3(BS / 128, D_DIM / 128, 3), blk, 0, stream>>>(
        query, keys, values, Wq, Wk, Wv, bq, bk, bv, Qw, Kw, Vw);
    attn_kernel<<<dim3(B_SZ * H_NUM, S_LEN / 128), blk, 0, stream>>>(Qw, Kw, Vw, Att);
    out_proj_kernel<<<dim3(BS / 128, D_DIM / 128), blk, 0, stream>>>(Att, Wo, bo, out);
}
